// MSMTTEGRUAtt64Model_2345052144360
// MI455X (gfx1250) — matrix operand order repaired and checked by reading, not compiled, not run on hardware
//
#include <hip/hip_runtime.h>
#include <hip/hip_bf16.h>
#include <math.h>

typedef __attribute__((ext_vector_type(16))) _Float16     v16h;
typedef __attribute__((ext_vector_type(8)))  float        v8f;
typedef __attribute__((ext_vector_type(4)))  float        v4f;
typedef __attribute__((ext_vector_type(4)))  unsigned int v4u;
typedef __attribute__((ext_vector_type(8)))  int          v8i;
typedef __attribute__((ext_vector_type(4)))  int          v4i;

#define BB   128
#define TT   1024
#define HH   64
#define G3H  192   // 3*H
#define CH   16    // timesteps per staged chunk
#define NCH  (TT / CH)

#if defined(__HIP_DEVICE_COMPILE__) && __has_builtin(__builtin_amdgcn_tensor_load_to_lds) && __has_builtin(__builtin_amdgcn_s_wait_tensorcnt)
#define HAVE_TDM 1
#else
#define HAVE_TDM 0
#endif

static __device__ __forceinline__ float rcp_fast(float x) {
#if __has_builtin(__builtin_amdgcn_rcpf)
    return __builtin_amdgcn_rcpf(x);
#else
    return 1.0f / x;
#endif
}
static __device__ __forceinline__ float sigmoid_fast(float x) {
    return rcp_fast(1.0f + __expf(-x));
}
static __device__ __forceinline__ float tanh_fast(float x) {
#if __has_builtin(__builtin_amdgcn_tanhf)
    return __builtin_amdgcn_tanhf(x);
#else
    return 1.0f - 2.0f * rcp_fast(1.0f + __expf(2.0f * x));
#endif
}

#if HAVE_TDM
// Issue one 2-D TDM tile load: 16 rows x tile_d0 4-byte elements, row stride stride0.
static __device__ __forceinline__ void tdm_load_2d(
    unsigned lds_addr, unsigned long long gaddr,
    unsigned tensor_d0, unsigned tile_d0, unsigned long long stride0)
{
    v4u g0 = { 1u,                                   // count=1 (valid descriptor)
               lds_addr,                             // lds_addr
               (unsigned)(gaddr & 0xFFFFFFFFull),    // global_addr[31:0]
               ((unsigned)((gaddr >> 32) & 0x01FFFFFFull)) | 0x80000000u }; // addr[56:32] | type=2
    v8i g1;
    g1[0] = (int)(2u << 16);                                        // data_size=4B, no multicast
    g1[1] = (int)((tensor_d0 & 0xFFFFu) << 16);                     // tensor_dim0[15:0]
    g1[2] = (int)(((tensor_d0 >> 16) & 0xFFFFu) | (16u << 16));     // tensor_dim0[31:16] | tensor_dim1[15:0]=16
    g1[3] = (int)((tile_d0 & 0xFFFFu) << 16);                       // tensor_dim1[31:16]=0 | tile_dim0
    g1[4] = (int)16;                                                // tile_dim1=16, tile_dim2=0
    g1[5] = (int)(stride0 & 0xFFFFFFFFull);                         // tensor_dim0_stride[31:0]
    g1[6] = (int)((stride0 >> 32) & 0xFFFFull);                     // stride[47:32] | dim1_stride[15:0]=0
    g1[7] = 0;
    v4i gz = { 0, 0, 0, 0 };
#if __clang_major__ >= 23
    v8i z8 = { 0, 0, 0, 0, 0, 0, 0, 0 };
    __builtin_amdgcn_tensor_load_to_lds(g0, g1, gz, gz, z8, 0);
#else
    __builtin_amdgcn_tensor_load_to_lds(g0, g1, gz, gz, 0);
#endif
}
#endif

// ---------------------------------------------------------------------------
// GRU recurrence body, fully specialized per GRU kind.
//   G: 0=spatial(DIN=2), 1=dist(DIN=8), 2=hour(LUT), 3=week(LUT)
// One workgroup = 16 batch rows; 12 waves, wave n owns output cols [16n,16n+16).
// ---------------------------------------------------------------------------
template<int G, int DIN>
static __device__ __forceinline__ void gru_body(
    const float* __restrict__ xin, const int* __restrict__ idxp,
    const float* __restrict__ emb, const int nemb,
    const float* __restrict__ kern, const float* __restrict__ rk, const float* __restrict__ bias,
    const int* __restrict__ seq_lens, float* __restrict__ h_out, const int slot,
    float* lut, _Float16* h16, float* hf, float* zs, float* rs,
    float* xchunk, int* ichunk)
{
    const int b0   = blockIdx.x * 16;
    const int tid  = threadIdx.x;
    const int wave = tid >> 5;
    const int lane = tid & 31;
    const int kq   = lane >> 4;
    const int lc   = lane & 15;
    const int c    = wave * 16 + lc;   // output column in [0,192)
    const int gate = wave >> 2;        // 0=z, 1=r, 2=h
    const int jl   = c - gate * 64;

    constexpr int BUFX = 16 * CH * DIN;   // floats per x chunk buffer
    constexpr int BUFI = 16 * CH;         // ints per idx chunk buffer

    // Input-projection LUT for embedding GRUs (includes b_in)
    if constexpr (G >= 2) {
        for (int i = tid; i < nemb * G3H; i += 384) {
            int e = i / G3H, cc = i % G3H;
            float s = bias[cc];
            #pragma unroll 8
            for (int k = 0; k < HH; ++k) s = fmaf(emb[e * HH + k], kern[k * G3H + cc], s);
            lut[i] = s;
        }
    }
    for (int i = tid; i < 16 * HH; i += 384) { hf[i] = 0.0f; h16[i] = (_Float16)0.0f; }

    // Loop-invariant per-lane registers
    const float brec_c = bias[G3H + c];
    float binc = 0.0f, ksc[DIN];
    if constexpr (G < 2) {
        binc = bias[c];
        #pragma unroll
        for (int d = 0; d < DIN; ++d) ksc[d] = kern[d * G3H + c];
    }
    int len_i[8];
    #pragma unroll
    for (int i = 0; i < 8; ++i) len_i[i] = seq_lens[b0 + kq * 8 + i];

    // B fragments (rkernel cols for this wave), f16, register-resident all T.
    v16h Bf0, Bf1;
    #pragma unroll
    for (int e = 0; e < 16; ++e) {
        Bf0[e] = (_Float16)rk[(kq * 8 + e + (e >= 8 ? 8 : 0))      * G3H + c];
        Bf1[e] = (_Float16)rk[(32 + kq * 8 + e + (e >= 8 ? 8 : 0)) * G3H + c];
    }

#if HAVE_TDM
    if (wave == 0) {   // prime chunk 0 into buffer 0
        if constexpr (G < 2)
            tdm_load_2d((unsigned)(size_t)(void*)xchunk,
                        (unsigned long long)(size_t)(const void*)&xin[(size_t)b0 * TT * DIN],
                        (unsigned)(TT * DIN), (unsigned)(CH * DIN),
                        (unsigned long long)(TT * DIN));
        else
            tdm_load_2d((unsigned)(size_t)(void*)ichunk,
                        (unsigned long long)(size_t)(const void*)&idxp[(size_t)b0 * TT],
                        (unsigned)TT, (unsigned)CH, (unsigned long long)TT);
    }
#endif
    __syncthreads();

    for (int tc = 0; tc < NCH; ++tc) {
        const int buf = tc & 1;
#if HAVE_TDM
        if (wave == 0) {
            if (tc + 1 < NCH) {   // prefetch next chunk into other buffer, then wait current
                const int t0 = (tc + 1) * CH;
                if constexpr (G < 2)
                    tdm_load_2d((unsigned)(size_t)(void*)(xchunk + (buf ^ 1) * BUFX),
                                (unsigned long long)(size_t)(const void*)&xin[((size_t)b0 * TT + t0) * DIN],
                                (unsigned)(TT * DIN), (unsigned)(CH * DIN),
                                (unsigned long long)(TT * DIN));
                else
                    tdm_load_2d((unsigned)(size_t)(void*)(ichunk + (buf ^ 1) * BUFI),
                                (unsigned long long)(size_t)(const void*)&idxp[(size_t)b0 * TT + t0],
                                (unsigned)TT, (unsigned)CH, (unsigned long long)TT);
                __builtin_amdgcn_s_wait_tensorcnt((short)1);
            } else {
                __builtin_amdgcn_s_wait_tensorcnt((short)0);
            }
        }
#else
        // Fallback: cooperative synchronous chunk load
        if constexpr (G < 2) {
            for (int i = tid; i < BUFX; i += 384) {
                int m = i / (CH * DIN), r = i % (CH * DIN);
                xchunk[buf * BUFX + i] = xin[((size_t)(b0 + m) * TT + tc * CH) * DIN + r];
            }
        } else {
            for (int i = tid; i < BUFI; i += 384) {
                int m = i / CH, r = i % CH;
                ichunk[buf * BUFI + i] = idxp[(size_t)(b0 + m) * TT + tc * CH + r];
            }
        }
#endif
        __syncthreads();

        for (int tt = 0; tt < CH; ++tt) {
            const int t = tc * CH + tt;

            // A fragments from h16 (16x32 f16 layout, two K slices)
            v16h a0, a1;
            {
                const _Float16* hb = h16 + lc * HH;
                ((v4f*)&a0)[0] = *(const v4f*)(hb + kq * 8);
                ((v4f*)&a0)[1] = *(const v4f*)(hb + 16 + kq * 8);
                ((v4f*)&a1)[0] = *(const v4f*)(hb + 32 + kq * 8);
                ((v4f*)&a1)[1] = *(const v4f*)(hb + 48 + kq * 8);
            }

            // rw tile = h @ rk[:,ctile] + b_rec
            v8f acc;
            #pragma unroll
            for (int i = 0; i < 8; ++i) acc[i] = brec_c;
            acc = __builtin_amdgcn_wmma_f32_16x16x32_f16(false, a0, false, Bf0, (short)0, acc, false, false);
            acc = __builtin_amdgcn_wmma_f32_16x16x32_f16(false, a1, false, Bf1, (short)0, acc, false, false);

            // input projection for the 8 (row, c) elements this lane owns
            float xv[8];
            #pragma unroll
            for (int i = 0; i < 8; ++i) {
                const int m = kq * 8 + i;
                if constexpr (G < 2) {
                    const float* xr = xchunk + buf * BUFX + m * (CH * DIN) + tt * DIN;
                    float s = binc;
                    #pragma unroll
                    for (int d = 0; d < DIN; ++d) s = fmaf(xr[d], ksc[d], s);
                    xv[i] = s;
                } else {
                    xv[i] = lut[ichunk[buf * BUFI + m * CH + tt] * G3H + c];
                }
            }

            if (gate == 0) {
                #pragma unroll
                for (int i = 0; i < 8; ++i) { int m = kq * 8 + i; zs[m * HH + jl] = sigmoid_fast(xv[i] + acc[i]); }
            } else if (gate == 1) {
                #pragma unroll
                for (int i = 0; i < 8; ++i) { int m = kq * 8 + i; rs[m * HH + jl] = sigmoid_fast(xv[i] + acc[i]); }
            }
            __syncthreads();

            if (gate == 2) {
                #pragma unroll
                for (int i = 0; i < 8; ++i) {
                    const int m = kq * 8 + i;
                    const float r  = rs[m * HH + jl];
                    const float z  = zs[m * HH + jl];
                    const float hp = hf[m * HH + jl];
                    const float hh = tanh_fast(xv[i] + r * acc[i]);
                    const float hn = z * hp + (1.0f - z) * hh;
                    const float res = (t < len_i[i]) ? hn : hp;
                    hf [m * HH + jl] = res;
                    h16[m * HH + jl] = (_Float16)res;
                }
            }
            __syncthreads();
        }
    }

    // final hidden -> h_out[slot][B][H]
    for (int i = tid; i < 16 * HH; i += 384) {
        int m = i >> 6, j = i & 63;
        h_out[((size_t)slot * BB + b0 + m) * HH + j] = hf[i];
    }
}

// ---------------------------------------------------------------------------
__global__ __launch_bounds__(384) void gru_kernel(
    const float* __restrict__ spatial, const float* __restrict__ dist,
    const int* __restrict__ hour_idx, const int* __restrict__ week_idx,
    const int* __restrict__ seq_lens,
    const float* __restrict__ time_emb, const float* __restrict__ week_emb,
    const float* __restrict__ k_sp, const float* __restrict__ rk_sp, const float* __restrict__ b_sp,
    const float* __restrict__ k_ds, const float* __restrict__ rk_ds, const float* __restrict__ b_ds,
    const float* __restrict__ k_hr, const float* __restrict__ rk_hr, const float* __restrict__ b_hr,
    const float* __restrict__ k_wk, const float* __restrict__ rk_wk, const float* __restrict__ b_wk,
    float* __restrict__ h_out)
{
    __shared__ __align__(16) float    lut[24 * G3H];
    __shared__ __align__(16) _Float16 h16[16 * HH];
    __shared__ __align__(16) float    hf [16 * HH];
    __shared__ __align__(16) float    zs [16 * HH];
    __shared__ __align__(16) float    rs [16 * HH];
    __shared__ __align__(16) float    xchunk[2 * 16 * CH * 8];
    __shared__ __align__(16) int      ichunk[2 * 16 * CH];

    const int g = blockIdx.y;
    if (g == 0)
        gru_body<0, 2>(spatial, nullptr, nullptr, 0, k_sp, rk_sp, b_sp, seq_lens, h_out, 0,
                       lut, h16, hf, zs, rs, xchunk, ichunk);
    else if (g == 1)
        gru_body<1, 8>(dist, nullptr, nullptr, 0, k_ds, rk_ds, b_ds, seq_lens, h_out, 3,
                       lut, h16, hf, zs, rs, xchunk, ichunk);
    else if (g == 2)
        gru_body<2, 1>(nullptr, hour_idx, time_emb, 24, k_hr, rk_hr, b_hr, seq_lens, h_out, 1,
                       lut, h16, hf, zs, rs, xchunk, ichunk);
    else
        gru_body<3, 1>(nullptr, week_idx, week_emb, 7, k_wk, rk_wk, b_wk, seq_lens, h_out, 2,
                       lut, h16, hf, zs, rs, xchunk, ichunk);
}

// ---------------------------------------------------------------------------
// Attention head: feats[B,64,4] -> softmax(relu(feats@Wa+ba), axis=2) -> scored
// ---------------------------------------------------------------------------
__global__ void attn_kernel(const float* __restrict__ h_out, const float* __restrict__ Wa,
                            const float* __restrict__ ba, float* __restrict__ scored)
{
    int i = blockIdx.x * blockDim.x + threadIdx.x;
    if (i >= BB * HH) return;
    int b = i >> 6, j = i & 63;
    float f[4];
    #pragma unroll
    for (int s = 0; s < 4; ++s) f[s] = h_out[((size_t)s * BB + b) * HH + j];
    float sc[4], mx = -1e30f;
    #pragma unroll
    for (int k = 0; k < 4; ++k) {
        float v = ba[k];
        #pragma unroll
        for (int s = 0; s < 4; ++s) v = fmaf(f[s], Wa[s * 4 + k], v);
        v = fmaxf(v, 0.0f);
        sc[k] = v; mx = fmaxf(mx, v);
    }
    float den = 0.0f;
    #pragma unroll
    for (int k = 0; k < 4; ++k) { sc[k] = __expf(sc[k] - mx); den += sc[k]; }
    float o = 0.0f;
    #pragma unroll
    for (int k = 0; k < 4; ++k) o = fmaf(f[k], sc[k], o);
    scored[i] = o * rcp_fast(den);
}

// ---------------------------------------------------------------------------
// Generic bias+ReLU GEMM via f16 WMMA: C[M,N] = act(A[M,K] @ W[K,N] + b)
// ---------------------------------------------------------------------------
__global__ __launch_bounds__(256) void gemm_relu16(
    const float* __restrict__ A, const float* __restrict__ W,
    const float* __restrict__ bias, float* __restrict__ C,
    int M, int K, int N, int do_relu)
{
    const int lane = threadIdx.x & 31;
    const int waveId = blockIdx.x * 8 + (threadIdx.x >> 5);
    const int ntilesN = N >> 4;
    const int ntiles  = (M >> 4) * ntilesN;
    if (waveId >= ntiles) return;
    const int mt = waveId / ntilesN;
    const int nt = waveId % ntilesN;
    const int kq = lane >> 4, lc = lane & 15;
    const int col  = nt * 16 + lc;
    const int rowA = mt * 16 + lc;

    v8f acc;
    const float bc = bias[col];
    #pragma unroll
    for (int i = 0; i < 8; ++i) acc[i] = bc;

    for (int k0 = 0; k0 < K; k0 += 32) {
        v16h a, b;
        #pragma unroll
        for (int e = 0; e < 16; ++e) {
            int ka = k0 + ((e < 8) ? (kq * 8 + e) : (16 + kq * 8 + (e - 8)));
            a[e] = (_Float16)A[(size_t)rowA * K + ka];
            int kb = k0 + kq * 8 + e + (e >= 8 ? 8 : 0);
            b[e] = (_Float16)W[(size_t)kb * N + col];
        }
        acc = __builtin_amdgcn_wmma_f32_16x16x32_f16(false, a, false, b, (short)0, acc, false, false);
    }

    #pragma unroll
    for (int i = 0; i < 8; ++i) {
        int m = mt * 16 + kq * 8 + i;
        float v = acc[i];
        if (do_relu) v = fmaxf(v, 0.0f);
        C[(size_t)m * N + col] = v;
    }
}

// ---------------------------------------------------------------------------
__global__ void final_kernel(const float* __restrict__ x4, const float* __restrict__ scored,
                             const float* __restrict__ Wf, const float* __restrict__ bf,
                             float* __restrict__ out)
{
    int b = blockIdx.x * blockDim.x + threadIdx.x;
    if (b >= BB) return;
    float acc = bf[0];
    #pragma unroll 16
    for (int j = 0; j < HH; ++j)
        acc = fmaf(x4[b * HH + j] + scored[b * HH + j], Wf[j], acc);
    out[b] = acc;
}

// ---------------------------------------------------------------------------
extern "C" void kernel_launch(void* const* d_in, const int* in_sizes, int n_in,
                              void* d_out, int out_size, void* d_ws, size_t ws_size,
                              hipStream_t stream)
{
    const float* spatial  = (const float*)d_in[0];
    const float* dist     = (const float*)d_in[1];
    const int*   hour_idx = (const int*)  d_in[2];
    const int*   week_idx = (const int*)  d_in[3];
    const int*   seq_lens = (const int*)  d_in[4];
    const float* time_emb = (const float*)d_in[5];
    const float* week_emb = (const float*)d_in[6];
    const float* k_sp  = (const float*)d_in[7];
    const float* rk_sp = (const float*)d_in[8];
    const float* b_sp  = (const float*)d_in[9];
    const float* k_ds  = (const float*)d_in[10];
    const float* rk_ds = (const float*)d_in[11];
    const float* b_ds  = (const float*)d_in[12];
    const float* k_hr  = (const float*)d_in[13];
    const float* rk_hr = (const float*)d_in[14];
    const float* b_hr  = (const float*)d_in[15];
    const float* k_wk  = (const float*)d_in[16];
    const float* rk_wk = (const float*)d_in[17];
    const float* b_wk  = (const float*)d_in[18];
    const float* Wa = (const float*)d_in[19];
    const float* ba = (const float*)d_in[20];
    const float* W1 = (const float*)d_in[21];
    const float* b1 = (const float*)d_in[22];
    const float* W2 = (const float*)d_in[23];
    const float* b2 = (const float*)d_in[24];
    const float* W3 = (const float*)d_in[25];
    const float* b3 = (const float*)d_in[26];
    const float* W4 = (const float*)d_in[27];
    const float* b4 = (const float*)d_in[28];
    const float* Wf = (const float*)d_in[29];
    const float* bf = (const float*)d_in[30];

    float* ws = (float*)d_ws;
    float* h_out  = ws;                       // 4*128*64
    float* scored = h_out + 4 * BB * HH;      // 128*64
    float* x1 = scored + BB * HH;             // 128*1024
    float* x2 = x1 + BB * 1024;               // 128*512
    float* x3 = x2 + BB * 512;                // 128*256
    float* x4 = x3 + BB * 256;                // 128*64

    gru_kernel<<<dim3(8, 4), 384, 0, stream>>>(
        spatial, dist, hour_idx, week_idx, seq_lens, time_emb, week_emb,
        k_sp, rk_sp, b_sp, k_ds, rk_ds, b_ds,
        k_hr, rk_hr, b_hr, k_wk, rk_wk, b_wk, h_out);

    attn_kernel<<<(BB * HH + 255) / 256, 256, 0, stream>>>(h_out, Wa, ba, scored);

    gemm_relu16<<<(8 * 64) / 8, 256, 0, stream>>>(scored, W1, b1, x1, BB, 64,   1024, 1);
    gemm_relu16<<<(8 * 32) / 8, 256, 0, stream>>>(x1,     W2, b2, x2, BB, 1024, 512,  1);
    gemm_relu16<<<(8 * 16) / 8, 256, 0, stream>>>(x2,     W3, b3, x3, BB, 512,  256,  1);
    gemm_relu16<<<(8 * 4)  / 8, 256, 0, stream>>>(x3,     W4, b4, x4, BB, 256,  64,   1);

    final_kernel<<<1, 128, 0, stream>>>(x4, scored, Wf, bf, (float*)d_out);
}
